// VectorQuantizer_27118423507212
// MI455X (gfx1250) — matrix operand order repaired and checked by reading, not compiled, not run on hardware
//
#include <hip/hip_runtime.h>

// CDNA5 / gfx1250 vector-quantizer (VQ-VAE) forward.
// score_k = 512*||e_k||^2 + z · (-1024*e_k)   (argmin-equivalent to L2 dist)
// computed with V_WMMA_F32_16X16X32_F16 using hi/lo f16 splitting of both
// operands for ~f32 accuracy. Two independent code tiles in flight per
// iteration; argmin compares are software-pipelined one iteration back so
// they fill the WMMA->VALU hazard slots. A-operands are pinned with an empty
// asm so the compiler cannot rematerialize the f16 split inside the loop.

typedef _Float16 v16h __attribute__((ext_vector_type(16)));
typedef float    v8f  __attribute__((ext_vector_type(8)));

#define N_VEC   65536          // B*H*W rows
#define DIMS    64             // embedding dim
#define K_CODES 1024           // codebook entries
#define ND_TOT  (N_VEC * DIMS) // total output elements of quantized tensor

#define WMMA_F16(a, b, c) \
  __builtin_amdgcn_wmma_f32_16x16x32_f16(false, (a), false, (b), (short)0, (c), false, false)

// ---------------------------------------------------------------------------
// Prep: convert codebook to scaled f16 hi/lo, compute scaled norms, zero loss.
// ---------------------------------------------------------------------------
__global__ __launch_bounds__(256) void vq_prep(const float* __restrict__ emb,
                                               _Float16* __restrict__ ehi,
                                               _Float16* __restrict__ elo,
                                               float* __restrict__ norms,
                                               float* __restrict__ lossAcc) {
  const int k = blockIdx.x * blockDim.x + threadIdx.x;
  if (k == 0) *lossAcc = 0.0f;
  if (k < K_CODES) {
    const float* e = emb + (size_t)k * DIMS;
    _Float16* ph = ehi + (size_t)k * DIMS;
    _Float16* pl = elo + (size_t)k * DIMS;
    float s = 0.0f;
#pragma unroll
    for (int j = 0; j < DIMS; ++j) {
      float v = e[j];
      s += v * v;
      float w = -1024.0f * v;          // fold (-2) and 512x argmin-invariant scale
      _Float16 hf = (_Float16)w;
      ph[j] = hf;
      pl[j] = (_Float16)(w - (float)hf);
    }
    norms[k] = 512.0f * s;             // scaled ||e_k||^2 (exact f32)
  }
}

// ---------------------------------------------------------------------------
// Main: per wave, one 16-row tile of z; loop over 32 pairs of code tiles.
// ---------------------------------------------------------------------------
__global__ __launch_bounds__(256) void vq_main(const float* __restrict__ z,
                                               const float* __restrict__ emb,
                                               const _Float16* __restrict__ ehi,
                                               const _Float16* __restrict__ elo,
                                               const float* __restrict__ norms,
                                               float* __restrict__ out,
                                               float* __restrict__ lossAcc) {
  __shared__ float sN[K_CODES];
  for (int i = threadIdx.x; i < K_CODES; i += 256) sN[i] = norms[i];
  __syncthreads();

  const int lane = threadIdx.x & 31;
  const int wave = threadIdx.x >> 5;
  const int hlf  = lane >> 4;          // 0: lanes 0-15, 1: lanes 16-31
  const int lrow = lane & 15;
  const int rowBase = (blockIdx.x * 8 + wave) * 16;

  // ---- A tile: load this wave's 16x64 f32 z tile in WMMA A layout, split ----
  // A-matrix 16x32 f16 layout: lane holds row (lane&15); elements 0..15 are
  // features k0 + hlf*16 + {0..15} (contiguous, matching B).
  const float* zr = z + (size_t)(rowBase + lrow) * DIMS;
  const int s0 = hlf * 16;
  float tv[32];
#pragma unroll
  for (int g = 0; g < 4; ++g) {        // feature groups 0,8,32,40 (+hlf*16)
    float4 a = *(const float4*)(zr + (g >> 1) * 32 + (g & 1) * 8 + s0);
    float4 b = *(const float4*)(zr + (g >> 1) * 32 + (g & 1) * 8 + s0 + 4);
    tv[8 * g + 0] = a.x; tv[8 * g + 1] = a.y; tv[8 * g + 2] = a.z; tv[8 * g + 3] = a.w;
    tv[8 * g + 4] = b.x; tv[8 * g + 5] = b.y; tv[8 * g + 6] = b.z; tv[8 * g + 7] = b.w;
  }
  v16h ah0, al0, ah1, al1;             // hi/lo for K-chunk 0 (feat 0-31) and 1 (32-63)
#pragma unroll
  for (int e = 0; e < 16; ++e) {
    { float v = tv[e];      _Float16 hf = (_Float16)v; ah0[e] = hf; al0[e] = (_Float16)(v - (float)hf); }
    { float v = tv[16 + e]; _Float16 hf = (_Float16)v; ah1[e] = hf; al1[e] = (_Float16)(v - (float)hf); }
  }
  // Pin the A operands: opaque to the optimizer -> no rematerialization of the
  // f16 split inside the K loop, and tv[] dies here.
  asm volatile("" : "+v"(ah0), "+v"(ah1), "+v"(al0), "+v"(al1));

  float bv[8];
  int   bi[8];
#pragma unroll
  for (int r = 0; r < 8; ++r) { bv[r] = __builtin_inff(); bi[r] = 0; }

  const v8f z8 = {0.f, 0.f, 0.f, 0.f, 0.f, 0.f, 0.f, 0.f};

  // ---- prologue: score pair 0 (codes lrow, 16+lrow) ----
  int   cdA = lrow, cdB = 16 + lrow;
  float nvA = sN[cdA], nvB = sN[cdB];
  v8f cA, cB;
  {
    const v16h* bhpA = (const v16h*)(ehi + ((size_t)cdA << 6));
    const v16h* blpA = (const v16h*)(elo + ((size_t)cdA << 6));
    const v16h* bhpB = (const v16h*)(ehi + ((size_t)cdB << 6));
    const v16h* blpB = (const v16h*)(elo + ((size_t)cdB << 6));
    v16h bhA0 = bhpA[hlf], bhA1 = bhpA[2 + hlf], blA0 = blpA[hlf], blA1 = blpA[2 + hlf];
    v16h bhB0 = bhpB[hlf], bhB1 = bhpB[2 + hlf], blB0 = blpB[hlf], blB1 = blpB[2 + hlf];
    cA = WMMA_F16(ah0, bhA0, z8);  cB = WMMA_F16(ah0, bhB0, z8);
    cA = WMMA_F16(ah1, bhA1, cA);  cB = WMMA_F16(ah1, bhB1, cB);
    cA = WMMA_F16(al0, bhA0, cA);  cB = WMMA_F16(al0, bhB0, cB);
    cA = WMMA_F16(al1, bhA1, cA);  cB = WMMA_F16(al1, bhB1, cB);
    cA = WMMA_F16(ah0, blA0, cA);  cB = WMMA_F16(ah0, blB0, cB);
    cA = WMMA_F16(ah1, blA1, cA);  cB = WMMA_F16(ah1, blB1, cB);
  }

  // ---- steady state: issue WMMAs for pair i, compare pair i-1 ----
#pragma clang loop unroll(disable)
  for (int n0 = 32; n0 < K_CODES; n0 += 32) {
    const int nA = n0 + lrow;
    const int nB = n0 + 16 + lrow;
    // B-matrix 32x16 f16 layout: lane holds code column (lane&15), 16
    // contiguous features starting at k0 + hlf*16.
    const v16h* bhpA = (const v16h*)(ehi + ((size_t)nA << 6));
    const v16h* blpA = (const v16h*)(elo + ((size_t)nA << 6));
    const v16h* bhpB = (const v16h*)(ehi + ((size_t)nB << 6));
    const v16h* blpB = (const v16h*)(elo + ((size_t)nB << 6));
    v16h bhA0 = bhpA[hlf], bhA1 = bhpA[2 + hlf], blA0 = blpA[hlf], blA1 = blpA[2 + hlf];
    v16h bhB0 = bhpB[hlf], bhB1 = bhpB[2 + hlf], blB0 = blpB[hlf], blB1 = blpB[2 + hlf];

    v8f dA, dB;
    dA = WMMA_F16(ah0, bhA0, z8);  dB = WMMA_F16(ah0, bhB0, z8);
    dA = WMMA_F16(ah1, bhA1, dA);  dB = WMMA_F16(ah1, bhB1, dB);
    dA = WMMA_F16(al0, bhA0, dA);  dB = WMMA_F16(al0, bhB0, dB);
    dA = WMMA_F16(al1, bhA1, dA);  dB = WMMA_F16(al1, bhB1, dB);
    dA = WMMA_F16(ah0, blA0, dA);  dB = WMMA_F16(ah0, blB0, dB);
    dA = WMMA_F16(ah1, blA1, dA);  dB = WMMA_F16(ah1, blB1, dB);

    // compare previous pair while the above WMMAs are in flight
#pragma unroll
    for (int r = 0; r < 8; ++r) {      // strict < keeps lowest index per lane
      float sa = cA[r] + nvA;
      if (sa < bv[r]) { bv[r] = sa; bi[r] = cdA; }
      float sb = cB[r] + nvB;
      if (sb < bv[r]) { bv[r] = sb; bi[r] = cdB; }
    }
    cA = dA; cB = dB;
    cdA = nA; cdB = nB;
    nvA = sN[nA]; nvB = sN[nB];
  }

  // ---- epilogue: compare last pair ----
#pragma unroll
  for (int r = 0; r < 8; ++r) {
    float sa = cA[r] + nvA;
    if (sa < bv[r]) { bv[r] = sa; bi[r] = cdA; }
    float sb = cB[r] + nvB;
    if (sb < bv[r]) { bv[r] = sb; bi[r] = cdB; }
  }

  // ---- argmin reduction within each 16-lane half (rows r / r+8) ----
#pragma unroll
  for (int off = 1; off < 16; off <<= 1) {
#pragma unroll
    for (int r = 0; r < 8; ++r) {
      float ov = __shfl_xor(bv[r], off, 32);
      int   oi = __shfl_xor(bi[r], off, 32);
      if (ov < bv[r] || (ov == bv[r] && oi < bi[r])) { bv[r] = ov; bi[r] = oi; }
    }
  }

  // ---- gather winning embeddings, write quantized, accumulate loss ----
  float acc = 0.0f;
#pragma unroll
  for (int r = 0; r < 8; ++r) {
    const int row = rowBase + hlf * 8 + r;
    const int idx = bi[r];
    const float4 e4 = *(const float4*)(emb + ((size_t)idx << 6) + lrow * 4);
    const float4 z4 = *(const float4*)(z   + ((size_t)row << 6) + lrow * 4);
    *(float4*)(out + ((size_t)row << 6) + lrow * 4) = e4;
    float dx = e4.x - z4.x, dy = e4.y - z4.y, dz = e4.z - z4.z, dw = e4.w - z4.w;
    acc += dx * dx + dy * dy + dz * dz + dw * dw;
  }
#pragma unroll
  for (int off = 1; off < 32; off <<= 1) acc += __shfl_xor(acc, off, 32);
  if (lane == 0) atomicAdd(lossAcc, acc);
}

// ---------------------------------------------------------------------------
// Finalize the two scalar losses.
// ---------------------------------------------------------------------------
__global__ void vq_final(const float* __restrict__ lossAcc, float* __restrict__ out) {
  const float S = *lossAcc;
  const float inv = 1.0f / (float)ND_TOT;
  out[ND_TOT]     = 0.25f * S * inv;   // commitment_loss
  out[ND_TOT + 1] = S * inv;           // codebook_loss
}

// ---------------------------------------------------------------------------
extern "C" void kernel_launch(void* const* d_in, const int* in_sizes, int n_in,
                              void* d_out, int out_size, void* d_ws, size_t ws_size,
                              hipStream_t stream) {
  const float* z   = (const float*)d_in[0];   // [16,64,64,64] f32
  const float* emb = (const float*)d_in[1];   // [1024,64] f32
  float* out = (float*)d_out;                 // [ND_TOT + 2] f32

  char* ws = (char*)d_ws;
  _Float16* ehi  = (_Float16*)(ws);            // 1024*64*2 = 131072 B
  _Float16* elo  = (_Float16*)(ws + 131072);   // 131072 B
  float*    nrm  = (float*)(ws + 262144);      // 4096 B
  float*    loss = (float*)(ws + 266240);      // 4 B

  vq_prep<<<4, 256, 0, stream>>>(emb, ehi, elo, nrm, loss);
  vq_main<<<512, 256, 0, stream>>>(z, emb, ehi, elo, nrm, out, loss);
  vq_final<<<1, 1, 0, stream>>>(loss, out);
}
